// Model3DETR_SepView_42013370090291
// MI455X (gfx1250) — compile-verified, matrix operand order checked by reading, not run on hardware
//
#include <hip/hip_runtime.h>
#include <hip/hip_bf16.h>

#define BATCH 64
#define NPTS  32768
#define NQ    256
#define DM    256
#define DH    128

typedef _Float16 v16h __attribute__((ext_vector_type(16)));
typedef _Float16 v8h  __attribute__((ext_vector_type(8)));
typedef float    v8f  __attribute__((ext_vector_type(8)));

// ---------------------------------------------------------------------------
// Kernel 1: Furthest point sampling. One block per batch, 1024 threads (32
// waves, wave32). Running min-distance kept in registers (32 per thread),
// xyz re-read from L2 each round. Argmax = (max val, min idx) reduction.
// ---------------------------------------------------------------------------
__global__ __launch_bounds__(1024) void fps_kernel(const float* __restrict__ xyz,
                                                   int* __restrict__ inds)
{
    const int b    = blockIdx.x;
    const int tid  = threadIdx.x;
    const int lane = tid & 31;
    const int wid  = tid >> 5;
    const float* bx = xyz + (size_t)b * NPTS * 3;

    __shared__ float s_val[32];
    __shared__ int   s_idx[32];
    __shared__ int   s_far;

    float dist[NPTS / 1024];
    #pragma unroll
    for (int i = 0; i < NPTS / 1024; ++i) dist[i] = 1e10f;

    int far = 0;
    if (tid == 0) inds[b * NQ + 0] = 0;

    for (int k = 1; k < NQ; ++k) {
        const float cx = bx[far * 3 + 0];
        const float cy = bx[far * 3 + 1];
        const float cz = bx[far * 3 + 2];

        float bv = -1.0f;
        int   bi = 0x7fffffff;
        #pragma unroll 4
        for (int i = 0; i < NPTS / 1024; ++i) {
            const int p = tid + i * 1024;
            const float dx = bx[p * 3 + 0] - cx;
            const float dy = bx[p * 3 + 1] - cy;
            const float dz = bx[p * 3 + 2] - cz;
            const float d  = dx * dx + dy * dy + dz * dz;
            const float nd = fminf(dist[i], d);
            dist[i] = nd;
            if (nd > bv || (nd == bv && p < bi)) { bv = nd; bi = p; }
        }
        // wave32 reduction: max value, first (smallest) index on ties
        #pragma unroll
        for (int off = 16; off > 0; off >>= 1) {
            const float ov = __shfl_down(bv, off);
            const int   oi = __shfl_down(bi, off);
            if (ov > bv || (ov == bv && oi < bi)) { bv = ov; bi = oi; }
        }
        if (lane == 0) { s_val[wid] = bv; s_idx[wid] = bi; }
        __syncthreads();
        if (wid == 0) {
            bv = s_val[lane];
            bi = s_idx[lane];
            #pragma unroll
            for (int off = 16; off > 0; off >>= 1) {
                const float ov = __shfl_down(bv, off);
                const int   oi = __shfl_down(bi, off);
                if (ov > bv || (ov == bv && oi < bi)) { bv = ov; bi = oi; }
            }
            if (lane == 0) s_far = bi;
        }
        __syncthreads();
        far = s_far;
        if (tid == 0) inds[b * NQ + k] = far;
    }
}

// ---------------------------------------------------------------------------
// Kernel 2: convert W1/W2 to f16 (shared across all batches, L2-resident).
// ---------------------------------------------------------------------------
__global__ void cvt_kernel(const float* __restrict__ W1, const float* __restrict__ W2,
                           _Float16* __restrict__ W1h, _Float16* __restrict__ W2h)
{
    const int i = blockIdx.x * blockDim.x + threadIdx.x;
    W1h[i] = (_Float16)W1[i];
    W2h[i] = (_Float16)W2[i];
}

// ---------------------------------------------------------------------------
// Kernel 3: gather query xyz + fourier embedding; write posT[b][n][c] (f16),
// i.e. pre-transposed so GEMM B-fragments are contiguous-K row loads.
// One block per (b, n), 128 threads (one per frequency).
// ---------------------------------------------------------------------------
__global__ __launch_bounds__(DH) void embed_kernel(const float* __restrict__ xyz,
                                                   const int* __restrict__ inds,
                                                   const float* __restrict__ pcmin,
                                                   const float* __restrict__ pcmax,
                                                   const float* __restrict__ gB,
                                                   float* __restrict__ qxyz,
                                                   _Float16* __restrict__ posT)
{
    const int b = blockIdx.x / NQ;
    const int n = blockIdx.x % NQ;
    const int f = threadIdx.x;           // 0..127

    const int idx = inds[b * NQ + n];
    const float qx = xyz[((size_t)b * NPTS + idx) * 3 + 0];
    const float qy = xyz[((size_t)b * NPTS + idx) * 3 + 1];
    const float qz = xyz[((size_t)b * NPTS + idx) * 3 + 2];
    if (f < 3) qxyz[(b * NQ + n) * 3 + f] = (f == 0 ? qx : (f == 1 ? qy : qz));

    const float nx = (qx - pcmin[b * 3 + 0]) / (pcmax[b * 3 + 0] - pcmin[b * 3 + 0]);
    const float ny = (qy - pcmin[b * 3 + 1]) / (pcmax[b * 3 + 1] - pcmin[b * 3 + 1]);
    const float nz = (qz - pcmin[b * 3 + 2]) / (pcmax[b * 3 + 2] - pcmin[b * 3 + 2]);

    const float proj = 6.283185307179586f *
        (nx * gB[f] + ny * gB[DH + f] + nz * gB[2 * DH + f]);

    _Float16* row = posT + ((size_t)b * NQ + n) * DM;
    row[f]      = (_Float16)sinf(proj);
    row[f + DH] = (_Float16)cosf(proj);
}

// ---------------------------------------------------------------------------
// Kernel 4: batched 256x256x256 GEMM via v_wmma_f32_16x16x32_f16.
// One wave = one 16x16 output tile, K-loop of 8 WMMAs.
//   A   = weights [DM][DM] f16, row-major (lane M = lane%16; two b128 K-chunks)
//   Bt  = activations [BATCH][NQ][DM] f16 (row n holds contiguous K)
//   outT != null : store relu(C+bias) as f16 into [BATCH][NQ][DM] (GEMM1 -> HT)
//   outE != null : store relu(C+bias) as f32 into [BATCH][DM][NQ] (GEMM2 -> d_out)
// ---------------------------------------------------------------------------
__global__ __launch_bounds__(256) void gemm_kernel(const _Float16* __restrict__ A,
                                                   const _Float16* __restrict__ Bt,
                                                   const float* __restrict__ bias,
                                                   _Float16* __restrict__ outT,
                                                   float* __restrict__ outE)
{
    const int wave = (blockIdx.x * blockDim.x + threadIdx.x) >> 5;
    const int lane = threadIdx.x & 31;
    const int b    = wave >> 8;         // 256 tiles per batch
    const int tile = wave & 255;
    const int tm   = tile >> 4;
    const int tn   = tile & 15;

    const int lmod  = lane & 15;
    const int khalf = (lane & 16) ? 8 : 0;   // lanes 16-31 hold K+8.. / K+24..

    const _Float16* arow = A  + (size_t)(tm * 16 + lmod) * DM + khalf;
    const _Float16* brow = Bt + ((size_t)b * NQ + (tn * 16 + lmod)) * DM + khalf;

    v8f c = {};
    #pragma unroll
    for (int k0 = 0; k0 < DM; k0 += 32) {
        const v8h alo = *(const v8h*)(arow + k0);
        const v8h ahi = *(const v8h*)(arow + k0 + 16);
        const v8h blo = *(const v8h*)(brow + k0);
        const v8h bhi = *(const v8h*)(brow + k0 + 16);
        const v16h a  = __builtin_shufflevector(alo, ahi,
                            0,1,2,3,4,5,6,7,8,9,10,11,12,13,14,15);
        const v16h bb = __builtin_shufflevector(blo, bhi,
                            0,1,2,3,4,5,6,7,8,9,10,11,12,13,14,15);
        c = __builtin_amdgcn_wmma_f32_16x16x32_f16(
                /*neg_a=*/false, a, /*neg_b=*/false, bb,
                /*c_mod=*/(short)0, c, /*reuse_a=*/false, /*reuse_b=*/false);
    }

    // C/D layout: lane -> N = lane%16; VGPR v -> M = tm*16 + (lane>=16 ? 8:0) + v
    const int dbase = tm * 16 + khalf;
    const int nn    = tn * 16 + lmod;

    if (outT) {
        v8h h;
        #pragma unroll
        for (int i = 0; i < 8; ++i) {
            const float r = c[i] + bias[dbase + i];
            h[i] = (_Float16)fmaxf(r, 0.0f);
        }
        *(v8h*)(outT + ((size_t)b * NQ + nn) * DM + dbase) = h;   // HT[n][d]
    } else {
        float* o = outE + (size_t)b * DM * NQ + nn;
        #pragma unroll
        for (int i = 0; i < 8; ++i) {
            const float r = c[i] + bias[dbase + i];
            o[(size_t)(dbase + i) * NQ] = fmaxf(r, 0.0f);         // E[d][n]
        }
    }
}

// ---------------------------------------------------------------------------
extern "C" void kernel_launch(void* const* d_in, const int* in_sizes, int n_in,
                              void* d_out, int out_size, void* d_ws, size_t ws_size,
                              hipStream_t stream)
{
    const float* xyz   = (const float*)d_in[0];  // [B,N,3]
    const float* pcmin = (const float*)d_in[1];  // [B,3]
    const float* pcmax = (const float*)d_in[2];  // [B,3]
    const float* gB    = (const float*)d_in[3];  // [3,128]
    const float* W1    = (const float*)d_in[4];  // [256,256]
    const float* b1    = (const float*)d_in[5];  // [256]
    const float* W2    = (const float*)d_in[6];  // [256,256]
    const float* b2    = (const float*)d_in[7];  // [256]

    char* ws = (char*)d_ws;
    int*      inds = (int*)ws;                               //   64*256*4   = 64 KB
    _Float16* W1h  = (_Float16*)(ws + (64 << 10));           //   256*256*2  = 128 KB
    _Float16* W2h  = (_Float16*)(ws + (192 << 10));          //   128 KB
    _Float16* posT = (_Float16*)(ws + (320 << 10));          //   64*256*256*2 = 8 MB
    _Float16* HT   = (_Float16*)(ws + (320 << 10) + (8u << 20)); // 8 MB

    float* qxyz = (float*)d_out;                             // [B,NQ,3]
    float* qemb = (float*)d_out + (size_t)BATCH * NQ * 3;    // [B,D,NQ]

    fps_kernel  <<<BATCH, 1024, 0, stream>>>(xyz, inds);
    cvt_kernel  <<<(DM * DM) / 256, 256, 0, stream>>>(W1, W2, W1h, W2h);
    embed_kernel<<<BATCH * NQ, DH, 0, stream>>>(xyz, inds, pcmin, pcmax, gB, qxyz, posT);

    // 64 batches * 256 tiles = 16384 waves; 8 waves per block
    gemm_kernel <<<2048, 256, 0, stream>>>(W1h, posT, b1, HT, nullptr);
    gemm_kernel <<<2048, 256, 0, stream>>>(W2h, HT, b2, nullptr, qemb);
}
